// Net1_128849019557
// MI455X (gfx1250) — matrix operand order repaired and checked by reading, not compiled, not run on hardware
//
#include <hip/hip_runtime.h>
#include <hip/hip_fp16.h>

typedef __attribute__((ext_vector_type(16))) _Float16 v16h;
typedef __attribute__((ext_vector_type(8)))  float    v8f;

#define D    128
#define NCLS 40

// ---------------- degree / norm ----------------
__global__ void k_deg(const int* __restrict__ dst, float* __restrict__ deg, int E) {
    int e = blockIdx.x * blockDim.x + threadIdx.x;
    if (e < E) atomicAdd(&deg[dst[e]], 1.0f);
}

__global__ void k_dinv(const float* __restrict__ deg, float* __restrict__ dinv, int n) {
    int i = blockIdx.x * blockDim.x + threadIdx.x;
    if (i < n) dinv[i] = rsqrtf(deg[i] + 1.0f);   // +1 for self loop
}

// ---------------- WMMA GEMM: C[N,128] = A[N,128] * W[128,128] ----------------
// One wave per 16-row slab. W is staged into LDS pre-swizzled into the exact
// B-fragment layout, so each lane's 16 f16 fragment elements are one 32-byte
// contiguous LDS load (2x ds_load_b128) instead of 16x ds_load_u16.
// Packed index: Wp[(k>>4)*2048 + col*16 + (k&15)] = W[k][col]
__global__ void __launch_bounds__(256) k_gemm(const float* __restrict__ A,
                                              const float* __restrict__ W,
                                              float* __restrict__ C, int n) {
    __shared__ _Float16 Wp[D * D];
    for (int i = threadIdx.x; i < D * D; i += 256) {
        const int k = i >> 7, col = i & 127;       // consecutive threads -> consecutive cols (coalesced)
        Wp[((k >> 4) * D + col) * 16 + (k & 15)] = (_Float16)W[i];
    }
    __syncthreads();

    const int wv = threadIdx.x >> 5;
    const int lane = threadIdx.x & 31;
    const int rowBase = (blockIdx.x * 8 + wv) * 16;
    if (rowBase >= n) return;                 // wave-uniform; N is a multiple of 16

    const int half = lane >> 4;               // which half-wave
    const int l15  = lane & 15;

    // A fragments for all 4 K-steps (contiguous k order, same as the packed B):
    // lanes 0-15: K = {0..15}; lanes 16-31: K = {16..31}
    const float4* ap4 = (const float4*)(A + (size_t)(rowBase + l15) * D);
    v16h af[4];
#pragma unroll
    for (int kk = 0; kk < 4; ++kk) {
        const int q = kk * 8 + half * 4;          // float4 index of K = kk*32 + half*16
        const float4 a0 = ap4[q + 0];
        const float4 a1 = ap4[q + 1];
        const float4 a2 = ap4[q + 2];
        const float4 a3 = ap4[q + 3];
        af[kk][0]  = (_Float16)a0.x; af[kk][1]  = (_Float16)a0.y;
        af[kk][2]  = (_Float16)a0.z; af[kk][3]  = (_Float16)a0.w;
        af[kk][4]  = (_Float16)a1.x; af[kk][5]  = (_Float16)a1.y;
        af[kk][6]  = (_Float16)a1.z; af[kk][7]  = (_Float16)a1.w;
        af[kk][8]  = (_Float16)a2.x; af[kk][9]  = (_Float16)a2.y;
        af[kk][10] = (_Float16)a2.z; af[kk][11] = (_Float16)a2.w;
        af[kk][12] = (_Float16)a3.x; af[kk][13] = (_Float16)a3.y;
        af[kk][14] = (_Float16)a3.z; af[kk][15] = (_Float16)a3.w;
    }

#pragma unroll
    for (int nt = 0; nt < 8; ++nt) {
        const int col = nt * 16 + l15;
        v8f acc = {};
#pragma unroll
        for (int kk = 0; kk < 4; ++kk) {
            // B 32x16 fragment: lane needs K = kk*32 + half*16 + 0..15 of column `col`,
            // which is 32 contiguous, 32B-aligned bytes in the packed layout.
            const v16h bf = *(const v16h*)(Wp + ((kk * 2 + half) * D + col) * 16);
            acc = __builtin_amdgcn_wmma_f32_16x16x32_f16(
                false, af[kk], false, bf, (short)0, acc, false, false);
        }
        // C layout: VGPR r -> row rowBase + r (+8 for upper half-wave), col = nt*16 + l15
#pragma unroll
        for (int r = 0; r < 8; ++r)
            C[(size_t)(rowBase + r + half * 8) * D + col] = acc[r];
    }
}

// ---------------- edge scatter: out[dst] += h[src] * dinv[src]*dinv[dst] ----------------
// One wave per edge; lane handles a float4 of channels.
__global__ void __launch_bounds__(256) k_scatter(const float* __restrict__ h,
                                                 const int* __restrict__ src,
                                                 const int* __restrict__ dst,
                                                 const float* __restrict__ dinv,
                                                 float* __restrict__ out, int E) {
    const int w = (int)((blockIdx.x * (unsigned)blockDim.x + threadIdx.x) >> 5);
    const int lane = threadIdx.x & 31;
    if (w >= E) return;
    const int s = src[w], d = dst[w];
    const float wt = dinv[s] * dinv[d];
    const float4 v = ((const float4*)(h + (size_t)s * D))[lane];
    float* o = out + (size_t)d * D + lane * 4;
    atomicAdd(o + 0, v.x * wt);
    atomicAdd(o + 1, v.y * wt);
    atomicAdd(o + 2, v.z * wt);
    atomicAdd(o + 3, v.w * wt);
}

// ---------------- self-loop + bias + ELU (in place on agg) ----------------
__global__ void k_epi(float* __restrict__ agg, const float* __restrict__ hmid,
                      const float* __restrict__ dinv, const float* __restrict__ b, int n) {
    const int idx = blockIdx.x * blockDim.x + threadIdx.x;
    if (idx >= n * D) return;
    const int node = idx >> 7, c = idx & 127;
    const float di = dinv[node];
    const float v = agg[idx] + hmid[idx] * di * di + b[c];
    agg[idx] = v > 0.f ? v : expm1f(v);       // ELU, alpha = 1
}

// ---------------- final linear (128->40) + log_softmax, one wave per node ----------------
__global__ void __launch_bounds__(256) k_final(const float* __restrict__ h,
                                               const float* __restrict__ Wl,
                                               const float* __restrict__ bl,
                                               float* __restrict__ out, int n) {
    __shared__ float sW[D * NCLS];
    __shared__ float sb[NCLS];
    for (int i = threadIdx.x; i < D * NCLS; i += blockDim.x) sW[i] = Wl[i];
    if (threadIdx.x < NCLS) sb[threadIdx.x] = bl[threadIdx.x];
    __syncthreads();

    const int wv = threadIdx.x >> 5;
    const int lane = threadIdx.x & 31;
    const int node = blockIdx.x * 8 + wv;
    if (node >= n) return;

    const float4 hv = *(const float4*)(h + (size_t)node * D + lane * 4);
    float l0 = 0.f;
    float l1 = -__builtin_inff();             // lanes 0..7 own classes 32..39

    for (int c = 0; c < NCLS; ++c) {
        const int k = lane * 4;
        float p = hv.x * sW[(k + 0) * NCLS + c] + hv.y * sW[(k + 1) * NCLS + c] +
                  hv.z * sW[(k + 2) * NCLS + c] + hv.w * sW[(k + 3) * NCLS + c];
#pragma unroll
        for (int off = 16; off; off >>= 1) p += __shfl_xor(p, off, 32);
        p += sb[c];
        if (c < 32) { if (lane == c) l0 = p; }
        else        { if (lane == c - 32) l1 = p; }
    }
    float m = fmaxf(l0, l1);
#pragma unroll
    for (int off = 16; off; off >>= 1) m = fmaxf(m, __shfl_xor(m, off, 32));
    float s = expf(l0 - m) + (lane < (NCLS - 32) ? expf(l1 - m) : 0.f);
#pragma unroll
    for (int off = 16; off; off >>= 1) s += __shfl_xor(s, off, 32);
    const float lse = m + logf(s);

    out[(size_t)node * NCLS + lane] = l0 - lse;
    if (lane < NCLS - 32) out[(size_t)node * NCLS + 32 + lane] = l1 - lse;
}

extern "C" void kernel_launch(void* const* d_in, const int* in_sizes, int n_in,
                              void* d_out, int out_size, void* d_ws, size_t ws_size,
                              hipStream_t stream) {
    const float* x  = (const float*)d_in[0];
    const int*   ei = (const int*)d_in[1];   // [2, E] flat: row0 = src, row1 = dst
    const float* W0 = (const float*)d_in[2];
    const float* b0 = (const float*)d_in[3];
    const float* W1 = (const float*)d_in[4];
    const float* b1 = (const float*)d_in[5];
    const float* W2 = (const float*)d_in[6];
    const float* b2 = (const float*)d_in[7];
    const float* Wl = (const float*)d_in[8];
    const float* bl = (const float*)d_in[9];

    const int N = in_sizes[0] / D;            // 50000
    const int E = in_sizes[1] / 2;            // 1600000
    const int* srcp = ei;
    const int* dstp = ei + E;

    float* ws   = (float*)d_ws;
    float* deg  = ws;
    float* dinv = ws + N;
    float* bufA = ws + 2 * (size_t)N;         // GEMM output (messages)
    float* bufB = bufA + (size_t)N * D;       // aggregation output / next input

    hipMemsetAsync(deg, 0, (size_t)N * sizeof(float), stream);
    k_deg<<<(E + 255) / 256, 256, 0, stream>>>(dstp, deg, E);
    k_dinv<<<(N + 255) / 256, 256, 0, stream>>>(deg, dinv, N);

    const int gemmBlocks = (N / 16 + 7) / 8;
    const int scatBlocks = (E + 7) / 8;       // 8 waves / 256-thread block
    const int epiBlocks  = (N * D + 255) / 256;

    const float* in = x;
    const float* Ws[3] = {W0, W1, W2};
    const float* bs[3] = {b0, b1, b2};
    for (int l = 0; l < 3; ++l) {
        k_gemm<<<gemmBlocks, 256, 0, stream>>>(in, Ws[l], bufA, N);
        hipMemsetAsync(bufB, 0, (size_t)N * D * sizeof(float), stream);
        k_scatter<<<scatBlocks, 256, 0, stream>>>(bufA, srcp, dstp, dinv, bufB, E);
        k_epi<<<epiBlocks, 256, 0, stream>>>(bufB, bufA, dinv, bs[l], N);
        in = bufB;
    }
    k_final<<<(N + 7) / 8, 256, 0, stream>>>(bufB, Wl, bl, (float*)d_out, N);
}
